// KANConv1d_65850438582447
// MI455X (gfx1250) — matrix operand order repaired and checked by reading, not compiled, not run on hardware
//
#include <hip/hip_runtime.h>

// ---------------------------------------------------------------------------
// KAN-Conv1d fused kernel for gfx1250 (MI455X), wave32 + WMMA bf16 + TDM.
//
// out[n, o] = sum_{i<576, g<9} A[n, i*9+g] * W[o, i*9+g]
//   g=0      : silu(x)           * base_weight[o, i]
//   g=1..8   : Bspline_{g-1}(x)  * spline_weight[o, i, g-1]
// where x = pf[n, i] = padded conv patch value, n = b*2048 + l.
//
// Per block: 128 rows x 128 cols, 8 waves as a 4(row) x 2(col) grid; each
// wave owns 2 row-tiles x 4 col-tiles (768 B LDS per WMMA vs 1152 before).
// K-chunks of 32 features (288 combined-K = 9 WMMA k-steps):
//   - TDM tensor_load_to_lds stages the W slice (overlapped with VALU work)
//   - threads zero the A-tile, then scatter silu + the 4 nonzero uniform
//     cubic B-spline bases per (row, feature) as bf16
//   - 9 k-steps x (2x4) v_wmma_f32_16x16x32_bf16 per wave
// Output is transposed through LDS for coalesced b128 stores along L.
// ---------------------------------------------------------------------------

typedef __attribute__((ext_vector_type(16))) __bf16 v16bf;
typedef __attribute__((ext_vector_type(8)))  float  v8f;
typedef __attribute__((ext_vector_type(4)))  unsigned int u32x4;
typedef __attribute__((ext_vector_type(8)))  int    i32x8;
typedef __attribute__((ext_vector_type(4)))  int    i32x4;

#define IN_CH    64
#define OUT_CH   128
#define KTAP     9
#define PADV     4
#define LLEN     2048
#define NBATCH   16
#define KAN_IN   576                 // IN_CH * KTAP
#define GPB      9                   // per-feature funcs: silu + 8 bases
#define KTOT     (KAN_IN * GPB)      // 5184 combined-K
#define M_TILE   128
#define FEAT_CHUNK 32
#define KCHUNK   (FEAT_CHUNK * GPB)  // 288 = 9 * 32
#define NCHUNKS  (KAN_IN / FEAT_CHUNK) // 18
#define A_STRIDE (KCHUNK + 8)        // 296 bf16: pad against LDS bank conflicts
#define C_STRIDE 132                 // f32 transpose stride (16B-aligned + pad)

#if defined(__AMDGCN__) && __has_builtin(__builtin_amdgcn_tensor_load_to_lds) && \
    __has_builtin(__builtin_amdgcn_s_wait_tensorcnt)
#define USE_TDM 1
#else
#define USE_TDM 0
#endif

struct bf16x16_bits { uint4 lo, hi; };  // 32B carrier for v16bf bit_cast

// --- weight prep: interleave base/spline weights into bf16 W[o][i*9+g] ------
__global__ __launch_bounds__(256)
void kan_prep_weights(const float* __restrict__ bw,   // (128, 576)
                      const float* __restrict__ sw,   // (128, 576, 8)
                      __bf16* __restrict__ W) {       // (128, 5184)
    int idx = blockIdx.x * 256 + threadIdx.x;
    if (idx >= OUT_CH * KTOT) return;
    int o  = idx / KTOT;
    int kk = idx - o * KTOT;
    int i  = kk / GPB;
    int g  = kk - i * GPB;
    float v = (g == 0) ? bw[o * KAN_IN + i]
                       : sw[(o * KAN_IN + i) * 8 + (g - 1)];
    W[o * KTOT + kk] = (__bf16)v;
}

// --- fused KAN conv ---------------------------------------------------------
__global__ __launch_bounds__(256)
void kan_conv_wmma(const float* __restrict__ x,      // (16, 64, 2048)
                   const __bf16* __restrict__ W,     // (128, 5184) bf16
                   float* __restrict__ out) {        // (16, 128, 2048)
    __shared__ __bf16 Alds[M_TILE * A_STRIDE];       // 75776 B (reused as Cl)
    __shared__ __bf16 Blds[OUT_CH * KCHUNK];         // 73728 B

    const int tid  = threadIdx.x;
    const int lane = tid & 31;
    const int wave = tid >> 5;
    const int wr   = wave & 3;                       // wave row-group (0..3)
    const int wc   = wave >> 2;                      // wave col-group (0..1)
    const int n0   = blockIdx.x * M_TILE;            // global row base
    const int b    = n0 / LLEN;                      // M_TILE | LLEN
    const int l0   = n0 - b * LLEN;

    v8f acc[2][4];
    const v8f vzero = {0.f, 0.f, 0.f, 0.f, 0.f, 0.f, 0.f, 0.f};
#pragma unroll
    for (int rt = 0; rt < 2; ++rt)
#pragma unroll
        for (int ct = 0; ct < 4; ++ct) acc[rt][ct] = vzero;

    for (int chunk = 0; chunk < NCHUNKS; ++chunk) {
        // ---- stage B-chunk: W[col][chunk*288 .. +288) -> Blds[col][0..288) --
#if USE_TDM
        if (tid == 0) {
            // Tensor DMA descriptor (D#): 2D tensor 5184 x 128 (2B elems),
            // tile 288 x 128 starting at byte offset chunk*288*2.
            unsigned long long ga =
                (unsigned long long)(const void*)W +
                (unsigned long long)chunk * (KCHUNK * 2);
            unsigned int ldsa = (unsigned int)(unsigned long long)(&Blds[0]);
            u32x4 g0 = {
                1u,                                        // count=1
                ldsa,                                      // lds_addr
                (unsigned int)ga,                          // global_addr lo
                (unsigned int)(ga >> 32) | (2u << 30)      // ga hi | type=2
            };
            i32x8 g1 = {
                (int)(1u << 16),                           // data_size=2B
                (int)((KTOT & 0xFFFF) << 16),              // tensor_dim0 lo16
                (int)((KTOT >> 16) | (OUT_CH << 16)),      // td0 hi | td1 lo
                (int)(KCHUNK << 16),                       // tile_dim0=288
                (int)OUT_CH,                               // tile_dim1=128
                (int)KTOT,                                 // td0_stride=5184
                0, 0
            };
            i32x4 gz4 = {0, 0, 0, 0};
            i32x8 gz8 = {0, 0, 0, 0, 0, 0, 0, 0};
            __builtin_amdgcn_tensor_load_to_lds(g0, g1, gz4, gz4, gz8, 0);
        }
#else
#pragma unroll 1
        for (int p = tid; p < OUT_CH * (KCHUNK / 8); p += 256) {
            int col = p / (KCHUNK / 8);
            int seg = p - col * (KCHUNK / 8);
            const uint4* src =
                (const uint4*)(W + col * KTOT + chunk * KCHUNK) + seg;
            ((uint4*)(Blds + col * KCHUNK))[seg] = *src;
        }
#endif

        // ---- zero A-tile (overlaps TDM) ------------------------------------
        {
            const uint4 z4 = {0u, 0u, 0u, 0u};
            uint4* Az = (uint4*)Alds;
#pragma unroll 1
            for (int p = tid; p < (M_TILE * A_STRIDE * 2) / 16; p += 256)
                Az[p] = z4;
        }
        __syncthreads();   // A-tile zero visible before scatter writes

        // ---- activations -> A-tile (overlaps TDM) --------------------------
#pragma unroll 1
        for (int p = 0; p < (M_TILE * FEAT_CHUNK) / 256; ++p) {
            int linear = p * 256 + tid;
            int il  = linear >> 7;          // feature-in-chunk (0..31)
            int row = linear & 127;         // row-in-tile; adj threads coalesce
            int i   = chunk * FEAT_CHUNK + il;
            int c   = i / KTAP;
            int tap = i - c * KTAP;
            int xpos = l0 + row + tap - PADV;
            float xv = 0.f;
            if (xpos >= 0 && xpos < LLEN)
                xv = x[(b * IN_CH + c) * LLEN + xpos];

            __bf16* arow = Alds + row * A_STRIDE + il * GPB;
            // silu (g = 0)
            float sg = 1.f / (1.f + __expf(-xv));
            arow[0] = (__bf16)(xv * sg);
            // uniform cubic B-spline: h=0.4, knots t_m = -2.2 + 0.4*m.
            // x in interval j -> nonzero bases m = j-3..j (clipped to [0,7]),
            // basis m = segment-(j-m) polynomial of the fractional part u.
            float s = (xv + 2.2f) * 2.5f;
            if (s >= 0.f && s < 11.f) {
                float fj = floorf(s);
                int   j  = (int)fj;
                float u  = s - fj;
                float u2 = u * u, u3 = u2 * u;
                const float c6 = 1.f / 6.f;
                float p0 = u3 * c6;                                   // d=0
                float p1 = (1.f + 3.f*u + 3.f*u2 - 3.f*u3) * c6;      // d=1
                float p2 = (4.f - 6.f*u2 + 3.f*u3) * c6;              // d=2
                float p3 = (1.f - 3.f*u + 3.f*u2 - u3) * c6;          // d=3
                if (j <= 7)            arow[1 + j]     = (__bf16)p0;
                if (j >= 1 && j <= 8)  arow[1 + j - 1] = (__bf16)p1;
                if (j >= 2 && j <= 9)  arow[1 + j - 2] = (__bf16)p2;
                if (j >= 3)            arow[1 + j - 3] = (__bf16)p3;
            }
        }
#if USE_TDM
        if (tid == 0) __builtin_amdgcn_s_wait_tensorcnt(0);
#endif
        __syncthreads();   // A scatter + B tile ready for everyone

        // ---- WMMA: 9 k-steps x (2 row-tiles x 4 col-tiles) per wave --------
        {
            const int arow0 = wr * 32 + (lane & 15);
            const int sel   = lane >> 4;   // half-wave select
#pragma unroll
            for (int ks = 0; ks < KCHUNK / 32; ++ks) {
                v16bf va[2];
#pragma unroll
                for (int rt = 0; rt < 2; ++rt) {
                    // A frag (16x32): lane<16 K {0..7,16..23}, lane>=16 +8
                    const __bf16* ap = Alds + (arow0 + rt * 16) * A_STRIDE
                                       + ks * 32 + sel * 8;
                    bf16x16_bits ab;
                    ab.lo = *(const uint4*)(ap);
                    ab.hi = *(const uint4*)(ap + 16);
                    va[rt] = __builtin_bit_cast(v16bf, ab);
                }
#pragma unroll
                for (int ct = 0; ct < 4; ++ct) {
                    // B frag (32x16): N = lane%16, K {0..7,16..23}, lane>=16 +8
                    const __bf16* bp =
                        Blds + (wc * 64 + ct * 16 + (lane & 15)) * KCHUNK
                        + ks * 32 + sel * 8;
                    bf16x16_bits bb;
                    bb.lo = *(const uint4*)(bp);
                    bb.hi = *(const uint4*)(bp + 16);
                    v16bf vb = __builtin_bit_cast(v16bf, bb);
#pragma unroll
                    for (int rt = 0; rt < 2; ++rt) {
                        acc[rt][ct] = __builtin_amdgcn_wmma_f32_16x16x32_bf16(
                            false, va[rt], false, vb, (short)0, acc[rt][ct],
                            false, false);
                    }
                }
            }
        }
        __syncthreads();   // done reading A/B before next chunk overwrites
    }

    // ---- writeback: transpose C through LDS for coalesced b128 stores ------
    {
        float* Cl = (float*)Alds;   // [col][C_STRIDE] = 128*132*4 = 67584 B
        const int mpart = (lane >> 4) << 3;   // +8 rows for upper half-wave
#pragma unroll
        for (int rt = 0; rt < 2; ++rt) {
#pragma unroll
            for (int ct = 0; ct < 4; ++ct) {
                int col   = wc * 64 + ct * 16 + (lane & 15);
                int rbase = wr * 32 + rt * 16 + mpart;
#pragma unroll
                for (int v = 0; v < 8; ++v)
                    Cl[col * C_STRIDE + rbase + v] = acc[rt][ct][v];
            }
        }
        __syncthreads();
#pragma unroll 1
        for (int u = tid; u < OUT_CH * (M_TILE / 4); u += 256) {
            int col  = u >> 5;          // 0..127
            int rseg = u & 31;          // 4-row segment
            float4 d = *(const float4*)(Cl + col * C_STRIDE + rseg * 4);
            *(float4*)(out + (b * OUT_CH + col) * LLEN + l0 + rseg * 4) = d;
        }
    }
}

// ---------------------------------------------------------------------------
extern "C" void kernel_launch(void* const* d_in, const int* in_sizes, int n_in,
                              void* d_out, int out_size, void* d_ws, size_t ws_size,
                              hipStream_t stream) {
    const float* x  = (const float*)d_in[0];  // (16, 64, 2048)
    const float* bw = (const float*)d_in[1];  // (128, 576)
    const float* sw = (const float*)d_in[2];  // (128, 576, 8)
    // d_in[3] = grid: uniform, baked into kernel constants.

    __bf16* W = (__bf16*)d_ws;                // 128*5184*2 = 1,327,104 B

    kan_prep_weights<<<(OUT_CH * KTOT + 255) / 256, 256, 0, stream>>>(bw, sw, W);

    const int nblocks = (NBATCH * LLEN) / M_TILE;   // 256
    kan_conv_wmma<<<nblocks, 256, 0, stream>>>(x, W, (float*)d_out);
}
